// GCN_7816840479101
// MI455X (gfx1250) — matrix operand order repaired and checked by reading, not compiled, not run on hardware
//
#include <hip/hip_runtime.h>

#define N_NODES  100000
#define N_EDGES  1600000
#define N_GRAPHS 512
#define IN_F     50
#define HID      128
#define OUT_F    4
#define KP1      64      // layer-1 K padded 50 -> 64 for f16 WMMA (K multiple of 32)

static_assert(N_NODES % 16 == 0, "GEMM row tiling requires N_NODES % 16 == 0");

typedef __attribute__((ext_vector_type(16))) _Float16 v16h;
typedef __attribute__((ext_vector_type(8)))  _Float16 v8h;
typedef __attribute__((ext_vector_type(4)))  _Float16 v4h;
typedef __attribute__((ext_vector_type(8)))  float    v8f;
typedef __attribute__((ext_vector_type(4)))  float    v4f;

// Native fp32 atomic add (relaxed, device scope) -> global_atomic_add_f32
__device__ __forceinline__ void atomic_add_f32(float* p, float v) {
    __hip_atomic_fetch_add(p, v, __ATOMIC_RELAXED, __HIP_MEMORY_SCOPE_AGENT);
}

// ---------------------------------------------------------------------------
// Utility kernels
// ---------------------------------------------------------------------------
__global__ void zero_f32v4(float* __restrict__ p, long long n4) {
    long long t = (long long)blockIdx.x * blockDim.x + threadIdx.x;
    if (t < n4) ((v4f*)p)[t] = (v4f){0.f, 0.f, 0.f, 0.f};
}

// f32 [rows, CIN] row-major -> f16 [rows, COUT] row-major, zero padded
template <int CIN, int COUT>
__global__ void cvt_pad_f16(const float* __restrict__ in, _Float16* __restrict__ out,
                            long long total) {
    long long t = (long long)blockIdx.x * blockDim.x + threadIdx.x;
    if (t >= total) return;
    int c = (int)(t % COUT);
    long long r = t / COUT;
    out[t] = (c < CIN) ? (_Float16)in[r * CIN + c] : (_Float16)0.0f;
}

// ---------------------------------------------------------------------------
// Edge scatter-add (segment_sum over dst of feat[src]) — f32 atomics to L2
// ---------------------------------------------------------------------------
// Layer 1: f32 features [N, FIN], accumulate into PADDED f32 agg [N, FPAD]
// (FPAD power of two -> index math is shift/mask, no integer division)
template <int FIN, int FPAD>
__global__ void scatter_add_f32_pad(const float* __restrict__ feat, const int* __restrict__ src,
                                    const int* __restrict__ dst, float* __restrict__ agg,
                                    long long total) {
    long long t = (long long)blockIdx.x * blockDim.x + threadIdx.x;
    if (t >= total) return;
    int f = (int)(t % FPAD);            // FPAD = 64 -> and
    long long e = t / FPAD;             // -> shift
    if (f >= FIN) return;
    int s = src[e], d = dst[e];
    atomic_add_f32(agg + (long long)d * FPAD + f, feat[(long long)s * FIN + f]);
}

// Layers 2/3: f16 features, 4 features per thread (one 8B load), f32 atomics
template <int F>
__global__ void scatter_add_f16v4(const _Float16* __restrict__ feat, const int* __restrict__ src,
                                  const int* __restrict__ dst, float* __restrict__ agg,
                                  long long total) {
    constexpr int FV = F / 4;
    long long t = (long long)blockIdx.x * blockDim.x + threadIdx.x;
    if (t >= total) return;
    int q = (int)(t % FV);              // FV = 32 -> and
    long long e = t / FV;               // -> shift
    int s = src[e], d = dst[e];
    v4h v = *(const v4h*)(feat + (long long)s * F + q * 4);
    float* base = agg + (long long)d * F + q * 4;
    atomic_add_f32(base + 0, (float)v[0]);
    atomic_add_f32(base + 1, (float)v[1]);
    atomic_add_f32(base + 2, (float)v[2]);
    atomic_add_f32(base + 3, (float)v[3]);
}

// ---------------------------------------------------------------------------
// Fused GraphConv GEMM:
//   out[n, :] = relu?( agg[n,:] @ Wrel^T + x[n,:] @ Wroot^T + b )
// A fragments: 16-bit A 16x32 ISA layout; B fragments: 16-bit B 32x16 layout.
// Block = 256 threads = 8 wave32; wave w -> 16x16 tile at cols [16w,16w+16).
// Requires N % 16 == 0 (guaranteed: N_NODES = 100000) -> no row guards,
// HAS32/HAS16 template bools -> straight-line unconditional store epilogue.
// ---------------------------------------------------------------------------
__device__ __forceinline__ v16h ldtile(const _Float16* p0, const _Float16* p1) {
    v8h lo = *(const v8h*)p0;
    v8h hh = *(const v8h*)p1;
    v16h r;
#pragma unroll
    for (int i = 0; i < 8; ++i) { r[i] = lo[i]; r[i + 8] = hh[i]; }
    return r;
}

template <int KP, int RELU, bool HAS32, bool HAS16>
__global__ __launch_bounds__(256) void gemm_graphconv(
    const _Float16* __restrict__ Aagg,   // [N, KP] f16
    const _Float16* __restrict__ Aroot,  // [N, KP] f16
    const _Float16* __restrict__ Wrel,   // [HID, KP] f16 (K contiguous per out-col)
    const _Float16* __restrict__ Wroot,  // [HID, KP] f16
    const float* __restrict__ bias,      // [HID]
    float* __restrict__ out32,           // [N, HID] f32 (used iff HAS32)
    _Float16* __restrict__ out16)        // [N, HID] f16 (used iff HAS16)
{
    const int lane  = threadIdx.x & 31;
    const int wave  = threadIdx.x >> 5;
    const int row0  = blockIdx.x << 4;
    const int colIn = lane & 15;
    const int col   = (wave << 4) + colIn;      // 8 waves * 16 = 128 output cols
    const int hi    = lane >> 4;                // lane half select

    // bias -> accumulator (bias depends only on col)
    v8f acc;
    const float bv = bias[col];
#pragma unroll
    for (int i = 0; i < 8; ++i) acc[i] = bv;

    // A: lane holds row m = row0 + (lane&15); low lanes K {0..7,16..23}, high {8..15,24..31}
    const int mA = row0 + colIn;
    const _Float16* pAg = Aagg  + (long long)mA * KP;
    const _Float16* pAr = Aroot + (long long)mA * KP;
    // B: lane holds col; low lanes K {0..7,16..23}, high {8..15,24..31}
    const _Float16* pBr = Wrel  + (long long)col * KP;
    const _Float16* pBo = Wroot + (long long)col * KP;

    const int aOff0 = hi ? 8  : 0;
    const int aOff1 = hi ? 24 : 16;
    const int bOff  = hi ? 16 : 0;

#pragma unroll
    for (int k0 = 0; k0 < KP; k0 += 32) {
        v16h a = ldtile(pAg + k0 + aOff0, pAg + k0 + aOff1);
        v16h b = ldtile(pBr + k0 + aOff0, pBr + k0 + aOff1);
        acc = __builtin_amdgcn_wmma_f32_16x16x32_f16(false, a, false, b,
                                                     (short)0, acc, false, false);
        v16h ar = ldtile(pAr + k0 + aOff0, pAr + k0 + aOff1);
        v16h br = ldtile(pBo + k0 + aOff0, pBo + k0 + aOff1);
        acc = __builtin_amdgcn_wmma_f32_16x16x32_f16(false, ar, false, br,
                                                     (short)0, acc, false, false);
    }

    // C/D layout: lanes 0-15 rows M=0..7 in acc[0..7]; lanes 16-31 rows M=8..15
    const long long obase = (long long)(row0 + (hi ? 8 : 0)) * HID + col;
#pragma unroll
    for (int r = 0; r < 8; ++r) {
        float v = acc[r];
        if (RELU) v = fmaxf(v, 0.0f);
        const long long o = obase + (long long)r * HID;
        if (HAS32) out32[o] = v;
        if (HAS16) out16[o] = (_Float16)v;
    }
}

// ---------------------------------------------------------------------------
// Global mean pool (accumulate) + final linear head
// ---------------------------------------------------------------------------
template <int F>
__global__ void pool_accum(const float* __restrict__ h, const int* __restrict__ batch,
                           float* __restrict__ pooled, float* __restrict__ counts,
                           long long total) {
    long long t = (long long)blockIdx.x * blockDim.x + threadIdx.x;
    if (t >= total) return;
    int f = (int)(t % F);               // F = 128 -> and
    long long n = t / F;                // -> shift
    int g = batch[n];
    atomic_add_f32(pooled + (long long)g * F + f, h[t]);
    if (f == 0) atomic_add_f32(counts + g, 1.0f);
}

__global__ void final_linear(const float* __restrict__ pooled, const float* __restrict__ counts,
                             const float* __restrict__ W, const float* __restrict__ b,
                             float* __restrict__ out, int G) {
    int t = blockIdx.x * blockDim.x + threadIdx.x;
    if (t >= G * OUT_F) return;
    int o = t % OUT_F;                  // OUT_F = 4 -> and
    int g = t / OUT_F;
    float inv = 1.0f / fmaxf(counts[g], 1.0f);
    const float* pg = pooled + (long long)g * HID;
    const float* wo = W + (long long)o * HID;
    float s = 0.0f;
#pragma unroll 4
    for (int k = 0; k < HID; ++k) s += pg[k] * wo[k];
    out[t] = s * inv + b[o];
}

// ---------------------------------------------------------------------------
// Host orchestration
// ---------------------------------------------------------------------------
static inline unsigned nblk(long long total, int bs) {
    return (unsigned)((total + bs - 1) / bs);
}

extern "C" void kernel_launch(void* const* d_in, const int* in_sizes, int n_in,
                              void* d_out, int out_size, void* d_ws, size_t ws_size,
                              hipStream_t stream) {
    const float* x       = (const float*)d_in[0];
    const int*   ei      = (const int*)  d_in[1];
    const int*   batch   = (const int*)  d_in[2];
    const float* W_rel1  = (const float*)d_in[3];
    const float* b_rel1  = (const float*)d_in[4];
    const float* W_root1 = (const float*)d_in[5];
    const float* W_rel2  = (const float*)d_in[6];
    const float* b_rel2  = (const float*)d_in[7];
    const float* W_root2 = (const float*)d_in[8];
    const float* W_rel3  = (const float*)d_in[9];
    const float* b_rel3  = (const float*)d_in[10];
    const float* W_root3 = (const float*)d_in[11];
    const float* W_lin   = (const float*)d_in[12];
    const float* b_lin   = (const float*)d_in[13];
    const int* src = ei;
    const int* dst = ei + N_EDGES;

    // workspace carve-up (256B aligned)
    char* p = (char*)d_ws;
    auto carve = [&](size_t bytes) -> char* {
        char* r = p;
        p += (bytes + 255) & ~(size_t)255;
        return r;
    };
    float*    aggF32 = (float*)   carve((size_t)N_NODES * HID * 4);  // also final h (aliased)
    _Float16* agg16  = (_Float16*)carve((size_t)N_NODES * HID * 2);
    _Float16* actA   = (_Float16*)carve((size_t)N_NODES * HID * 2);  // x16 (KP1) / layer-2 out
    _Float16* actB   = (_Float16*)carve((size_t)N_NODES * HID * 2);  // layer-1 out
    _Float16* w0     = (_Float16*)carve((size_t)HID * KP1 * 2);      // W_rel1  f16 padded
    _Float16* w1     = (_Float16*)carve((size_t)HID * KP1 * 2);      // W_root1
    _Float16* w2     = (_Float16*)carve((size_t)HID * HID * 2);      // W_rel2
    _Float16* w3     = (_Float16*)carve((size_t)HID * HID * 2);      // W_root2
    _Float16* w4     = (_Float16*)carve((size_t)HID * HID * 2);      // W_rel3
    _Float16* w5     = (_Float16*)carve((size_t)HID * HID * 2);      // W_root3
    float*    pooled = (float*)   carve((size_t)N_GRAPHS * HID * 4);
    float*    counts = (float*)   carve((size_t)N_GRAPHS * 4);
    float*    h32    = aggF32;   // layer-3 f32 output aliases agg buffer (agg copied to f16 first)

    const int BS = 256;
    const unsigned gemmBlocks = N_NODES / 16;

    // --- one-time conversions: x and weights -> f16 (padded K) ---
    cvt_pad_f16<IN_F, KP1><<<nblk((long long)N_NODES * KP1, BS), BS, 0, stream>>>(
        x, actA, (long long)N_NODES * KP1);
    cvt_pad_f16<IN_F, KP1><<<nblk((long long)HID * KP1, BS), BS, 0, stream>>>(
        W_rel1, w0, (long long)HID * KP1);
    cvt_pad_f16<IN_F, KP1><<<nblk((long long)HID * KP1, BS), BS, 0, stream>>>(
        W_root1, w1, (long long)HID * KP1);
    cvt_pad_f16<HID, HID><<<nblk((long long)HID * HID, BS), BS, 0, stream>>>(
        W_rel2, w2, (long long)HID * HID);
    cvt_pad_f16<HID, HID><<<nblk((long long)HID * HID, BS), BS, 0, stream>>>(
        W_root2, w3, (long long)HID * HID);
    cvt_pad_f16<HID, HID><<<nblk((long long)HID * HID, BS), BS, 0, stream>>>(
        W_rel3, w4, (long long)HID * HID);
    cvt_pad_f16<HID, HID><<<nblk((long long)HID * HID, BS), BS, 0, stream>>>(
        W_root3, w5, (long long)HID * HID);

    // --- Layer 1: aggregate in padded 64-wide input space (shift/mask indexing) ---
    zero_f32v4<<<nblk((long long)N_NODES * KP1 / 4, BS), BS, 0, stream>>>(
        aggF32, (long long)N_NODES * KP1 / 4);
    scatter_add_f32_pad<IN_F, KP1><<<nblk((long long)N_EDGES * KP1, BS), BS, 0, stream>>>(
        x, src, dst, aggF32, (long long)N_EDGES * KP1);
    cvt_pad_f16<KP1, KP1><<<nblk((long long)N_NODES * KP1, BS), BS, 0, stream>>>(
        aggF32, agg16, (long long)N_NODES * KP1);
    gemm_graphconv<KP1, 1, false, true><<<gemmBlocks, BS, 0, stream>>>(
        agg16, actA, w0, w1, b_rel1, nullptr, actB);

    // --- Layer 2 ---
    zero_f32v4<<<nblk((long long)N_NODES * HID / 4, BS), BS, 0, stream>>>(
        aggF32, (long long)N_NODES * HID / 4);
    scatter_add_f16v4<HID><<<nblk((long long)N_EDGES * (HID / 4), BS), BS, 0, stream>>>(
        actB, src, dst, aggF32, (long long)N_EDGES * (HID / 4));
    cvt_pad_f16<HID, HID><<<nblk((long long)N_NODES * HID, BS), BS, 0, stream>>>(
        aggF32, agg16, (long long)N_NODES * HID);
    gemm_graphconv<HID, 1, false, true><<<gemmBlocks, BS, 0, stream>>>(
        agg16, actB, w2, w3, b_rel2, nullptr, actA);

    // --- Layer 3 (no relu, f32 output for pooling) ---
    zero_f32v4<<<nblk((long long)N_NODES * HID / 4, BS), BS, 0, stream>>>(
        aggF32, (long long)N_NODES * HID / 4);
    scatter_add_f16v4<HID><<<nblk((long long)N_EDGES * (HID / 4), BS), BS, 0, stream>>>(
        actA, src, dst, aggF32, (long long)N_EDGES * (HID / 4));
    cvt_pad_f16<HID, HID><<<nblk((long long)N_NODES * HID, BS), BS, 0, stream>>>(
        aggF32, agg16, (long long)N_NODES * HID);
    gemm_graphconv<HID, 0, true, false><<<gemmBlocks, BS, 0, stream>>>(
        agg16, actA, w4, w5, b_rel3, h32, nullptr);

    // --- Global mean pool + linear head ---
    zero_f32v4<<<nblk((long long)N_GRAPHS * HID / 4, BS), BS, 0, stream>>>(
        pooled, (long long)N_GRAPHS * HID / 4);
    zero_f32v4<<<nblk((long long)N_GRAPHS / 4, BS), BS, 0, stream>>>(
        counts, (long long)N_GRAPHS / 4);
    pool_accum<HID><<<nblk((long long)N_NODES * HID, BS), BS, 0, stream>>>(
        h32, batch, pooled, counts, (long long)N_NODES * HID);
    final_linear<<<nblk((long long)N_GRAPHS * OUT_F, BS), BS, 0, stream>>>(
        pooled, counts, W_lin, b_lin, (float*)d_out, N_GRAPHS);
}
